// mGRU_62380105008298
// MI455X (gfx1250) — matrix operand order repaired and checked by reading, not compiled, not run on hardware
//
#include <hip/hip_runtime.h>
#include <stdint.h>

// mGRU fused: two bf16 WMMA GEMMs (M=16384,N=1024,K=2048) with async-LDS
// double buffering. GEMM1 epilogue: forget=sigmoid(.), writes forget(f32) and
// interim_h(bf16) directly into GEMM2's A buffer. GEMM2 epilogue: tanh + blend,
// writes both tuple copies of the output.

#define M_TOT 16384
#define K_TOT 2048
#define N_TOT 1024
#define BM 256            // block tile M
#define BN 128            // block tile N
#define BK 32             // k-tile (== WMMA K)
#define KT (K_TOT / BK)   // 64 k-tiles
#define LDSW 40           // halves per LDS row: 32 data + 8 pad (80B rows, 16B-aligned)

typedef __attribute__((ext_vector_type(16))) __bf16 v16bf;
typedef __attribute__((ext_vector_type(8)))  __bf16 v8bf;
typedef __attribute__((ext_vector_type(8)))  float  v8f;

__device__ __forceinline__ uint16_t f2bf(float f) {
  uint32_t u = __float_as_uint(f);
  u += 0x7FFFu + ((u >> 16) & 1u);   // round-to-nearest-even
  return (uint16_t)(u >> 16);
}

// CDNA5 async copy: 16B global -> LDS, tracked by ASYNCcnt
__device__ __forceinline__ void async_cp16(uint32_t lds_byte, const char* gaddr) {
  asm volatile("global_load_async_to_lds_b128 %0, %1, off"
               :: "v"(lds_byte), "v"(gaddr) : "memory");
}
__device__ __forceinline__ void wait_async_le6() {
  asm volatile("s_wait_asynccnt 6" ::: "memory");
}
__device__ __forceinline__ void wait_async_0() {
  asm volatile("s_wait_asynccnt 0" ::: "memory");
}

// ---- prep: A1 = bf16(cat(x,h)); A2 x-half = bf16(x) (h-half filled by GEMM1) ----
__global__ void pack_A(const float* __restrict__ x, const float* __restrict__ h,
                       uint16_t* __restrict__ A1, uint16_t* __restrict__ A2) {
  size_t i = (size_t)blockIdx.x * blockDim.x + threadIdx.x;  // pair index
  size_t e = i * 2;
  int row = (int)(e >> 11);
  int col = (int)(e & 2047);
  uint32_t pk;
  if (col < 1024) {
    float2 v = *(const float2*)(x + (size_t)row * 1024 + col);
    pk = (uint32_t)f2bf(v.x) | ((uint32_t)f2bf(v.y) << 16);
    *(uint32_t*)(A2 + e) = pk;       // GEMM2's x-half too
  } else {
    float2 v = *(const float2*)(h + (size_t)row * 1024 + (col - 1024));
    pk = (uint32_t)f2bf(v.x) | ((uint32_t)f2bf(v.y) << 16);
  }
  *(uint32_t*)(A1 + e) = pk;
}

__global__ void conv_bf16(const float* __restrict__ src, uint16_t* __restrict__ dst,
                          int npairs) {
  int i = blockIdx.x * blockDim.x + threadIdx.x;
  if (i >= npairs) return;
  float2 v = *(const float2*)(src + (size_t)i * 2);
  *(uint32_t*)(dst + (size_t)i * 2) =
      (uint32_t)f2bf(v.x) | ((uint32_t)f2bf(v.y) << 16);
}

// ---- fused GEMM: C[M,N] = A[M,K] * W[N,K]^T, bf16 in / f32 acc ----
template <int PHASE>
__global__ __launch_bounds__(256) void gemm_fused(
    const uint16_t* __restrict__ A,     // [M, K] bf16, stride 4096B
    const uint16_t* __restrict__ Wb,    // [N, K] bf16, stride 4096B
    const float* __restrict__ bias,
    const float* __restrict__ hsrc,
    float* __restrict__ forget,
    uint16_t* __restrict__ a2h,         // PHASE0: A2 + 1024 (row stride 2048 halves)
    float* __restrict__ outp) {
  __shared__ __align__(16) uint16_t tA[2][BM * LDSW];
  __shared__ __align__(16) uint16_t tB[2][BN * LDSW];

  const int tid  = threadIdx.x;
  const int lane = tid & 31;
  const int w    = tid >> 5;
  const int wm   = w & 3;        // 4 waves along M (64 rows each)
  const int wn   = w >> 2;       // 2 waves along N (64 cols each)
  const int l16  = lane & 15;
  const int hi   = lane >> 4;    // lane-half selects contiguous K half (0..15 / 16..31)

  const int m0 = blockIdx.y * BM;
  const int n0 = blockIdx.x * BN;

  uint32_t ldsA[2], ldsB[2];
  ldsA[0] = (uint32_t)(uintptr_t)&tA[0][0];
  ldsA[1] = (uint32_t)(uintptr_t)&tA[1][0];
  ldsB[0] = (uint32_t)(uintptr_t)&tB[0][0];
  ldsB[1] = (uint32_t)(uintptr_t)&tB[1][0];

  // Per-thread copy-chunk offsets (invariant over k): 16B chunks, 4 segs/row.
  // A tile: 256 rows * 4 = 1024 chunks ; B tile: 128 rows * 4 = 512 chunks.
  uint32_t aLds[4], bLds[2];
  size_t   aGbl[4], bGbl[2];
#pragma unroll
  for (int i = 0; i < 4; ++i) {
    int c = tid + i * 256, row = c >> 2, seg = c & 3;
    aLds[i] = (uint32_t)(row * (LDSW * 2) + seg * 16);
    aGbl[i] = (size_t)row * (K_TOT * 2) + (size_t)seg * 16;
  }
#pragma unroll
  for (int i = 0; i < 2; ++i) {
    int c = tid + i * 256, row = c >> 2, seg = c & 3;
    bLds[i] = (uint32_t)(row * (LDSW * 2) + seg * 16);
    bGbl[i] = (size_t)row * (K_TOT * 2) + (size_t)seg * 16;
  }
  const char* Abase = (const char*)A  + (size_t)m0 * (K_TOT * 2);
  const char* Wbase = (const char*)Wb + (size_t)n0 * (K_TOT * 2);

  v8f acc[4][4];
  const v8f vzero = {0.f, 0.f, 0.f, 0.f, 0.f, 0.f, 0.f, 0.f};
#pragma unroll
  for (int i = 0; i < 4; ++i)
#pragma unroll
    for (int j = 0; j < 4; ++j) acc[i][j] = vzero;

  auto issue = [&](int kt, int buf) {
    const size_t ko = (size_t)kt * (BK * 2);   // 64B per k-tile
#pragma unroll
    for (int i = 0; i < 4; ++i)
      async_cp16(ldsA[buf] + aLds[i], Abase + aGbl[i] + ko);
#pragma unroll
    for (int i = 0; i < 2; ++i)
      async_cp16(ldsB[buf] + bLds[i], Wbase + bGbl[i] + ko);
  };

  issue(0, 0);
  for (int kt = 0; kt < KT; ++kt) {
    const int buf = kt & 1;
    if (kt + 1 < KT) { issue(kt + 1, buf ^ 1); wait_async_le6(); }
    else             { wait_async_0(); }
    __syncthreads();   // all waves' tile parts landed in LDS

    v16bf bfrag[4], afrag[4];
#pragma unroll
    for (int nf = 0; nf < 4; ++nf) {
      // B 32x16: lane = N col, lane-half selects contiguous K 0..15 / 16..31
      const uint16_t* p = &tB[buf][(wn * 64 + nf * 16 + l16) * LDSW + hi * 16];
      union { v16bf v; v8bf h2[2]; } u;
      u.h2[0] = *reinterpret_cast<const v8bf*>(p);
      u.h2[1] = *reinterpret_cast<const v8bf*>(p + 8);
      bfrag[nf] = u.v;
    }
#pragma unroll
    for (int mf = 0; mf < 4; ++mf) {
      // A 16x32: lane = M row, lane-half selects contiguous K 0..15 / 16..31
      const uint16_t* p = &tA[buf][(wm * 64 + mf * 16 + l16) * LDSW + hi * 16];
      union { v16bf v; v8bf h2[2]; } u;
      u.h2[0] = *reinterpret_cast<const v8bf*>(p);
      u.h2[1] = *reinterpret_cast<const v8bf*>(p + 8);
      afrag[mf] = u.v;
    }
#pragma unroll
    for (int mf = 0; mf < 4; ++mf)
#pragma unroll
      for (int nf = 0; nf < 4; ++nf)
        acc[mf][nf] = __builtin_amdgcn_wmma_f32_16x16x32_bf16(
            false, afrag[mf], false, bfrag[nf], (short)0, acc[mf][nf], false, false);

    __syncthreads();   // protect buf from next-next tile's async writes
  }

  // Epilogue. C/D layout: VGPR r -> row r (lanes 0-15) / r+8 (lanes 16-31), N = lane%16.
  const int mw = m0 + wm * 64;
  const int nw = n0 + wn * 64;
#pragma unroll
  for (int mf = 0; mf < 4; ++mf) {
#pragma unroll
    for (int nf = 0; nf < 4; ++nf) {
#pragma unroll
      for (int r = 0; r < 8; ++r) {
        int m = mw + mf * 16 + hi * 8 + r;
        int n = nw + nf * 16 + l16;
        size_t idx = (size_t)m * N_TOT + n;
        float c = acc[mf][nf][r] + bias[n];
        if (PHASE == 0) {
          float f = 1.0f / (1.0f + __expf(-c));
          forget[idx] = f;
          a2h[(size_t)m * K_TOT + n] = f2bf(f * hsrc[idx]);  // into A2 h-half
        } else {
          float o = tanhf(c);
          float f = forget[idx];
          float hv = hsrc[idx];
          float res = (1.0f - f) * hv + f * o;
          outp[idx] = res;
          outp[(size_t)M_TOT * N_TOT + idx] = res;
        }
      }
    }
  }
}

extern "C" void kernel_launch(void* const* d_in, const int* in_sizes, int n_in,
                              void* d_out, int out_size, void* d_ws, size_t ws_size,
                              hipStream_t stream) {
  const float* x  = (const float*)d_in[0];
  const float* h  = (const float*)d_in[1];
  const float* W1 = (const float*)d_in[2];
  const float* b1 = (const float*)d_in[3];
  const float* W2 = (const float*)d_in[4];
  const float* b2 = (const float*)d_in[5];
  float* out = (float*)d_out;

  // workspace layout
  char* ws = (char*)d_ws;
  uint16_t* A1  = (uint16_t*)ws;                                    // 64 MB bf16 [M,2048]
  uint16_t* A2  = A1 + (size_t)M_TOT * K_TOT;                       // 64 MB bf16 [M,2048]
  uint16_t* W1b = A2 + (size_t)M_TOT * K_TOT;                       // 4 MB
  uint16_t* W2b = W1b + (size_t)N_TOT * K_TOT;                      // 4 MB
  float*    fgt = (float*)(W2b + (size_t)N_TOT * K_TOT);            // 64 MB f32 [M,1024]

  pack_A<<<dim3((M_TOT * (K_TOT / 2)) / 256), dim3(256), 0, stream>>>(x, h, A1, A2);
  conv_bf16<<<dim3((N_TOT * K_TOT / 2) / 256), dim3(256), 0, stream>>>(W1, W1b, N_TOT * K_TOT / 2);
  conv_bf16<<<dim3((N_TOT * K_TOT / 2) / 256), dim3(256), 0, stream>>>(W2, W2b, N_TOT * K_TOT / 2);

  dim3 grid(N_TOT / BN, M_TOT / BM);   // 8 x 64 = 512 blocks
  gemm_fused<0><<<grid, dim3(256), 0, stream>>>(
      A1, W1b, b1, h, fgt, A2 + (K_TOT / 2), out);
  gemm_fused<1><<<grid, dim3(256), 0, stream>>>(
      A2, W2b, b2, h, fgt, /*unused*/ A2 + (K_TOT / 2), out);
}
